// GrowableMoE_84284438216955
// MI455X (gfx1250) — matrix operand order repaired and checked by reading, not compiled, not run on hardware
//
#include <hip/hip_runtime.h>
#include <math.h>

typedef __bf16 bf16;
typedef __attribute__((ext_vector_type(16))) __bf16 v16bf;
typedef __attribute__((ext_vector_type(8)))  __bf16 v8bf;
typedef __attribute__((ext_vector_type(4)))  __bf16 v4bf;
typedef __attribute__((ext_vector_type(8)))  float  v8f;
typedef __attribute__((ext_vector_type(4)))  float  v4f;
typedef __attribute__((ext_vector_type(4)))  unsigned int u32x4;
typedef __attribute__((ext_vector_type(8)))  unsigned int u32x8;

#define T_TOK 8192   // B*S = 4*2048
#define HD    1024
#define ID    4096
#define NE    8

// ---------------------------------------------------------------- WMMA core
static __device__ __forceinline__ v8f wmma_bf16(v16bf a, v16bf b, v8f c) {
  // (neg_a, A, neg_b, B, c_mod, C, reuse_a, reuse_b)
  return __builtin_amdgcn_wmma_f32_16x16x32_bf16(false, a, false, b, (short)0, c,
                                                 false, false);
}

// A fragment (16x32 bf16): lane<16 holds row M=lane, K = {k0..k0+7, k0+16..k0+23}
//                          lane>=16 holds row M=lane-16, K = {k0+8..15, k0+24..31}
static __device__ __forceinline__ v16bf frag_a(const bf16* base, int ldk,
                                               int row0, int k0, int lane) {
  const bf16* p = base + (row0 + (lane & 15)) * ldk + k0 + ((lane & 16) ? 8 : 0);
  union { v16bf v; v8bf h[2]; } u;
  u.h[0] = *(const v8bf*)p;
  u.h[1] = *(const v8bf*)(p + 16);
  return u.v;
}

// B fragment (32x16 bf16): lane<16 holds col N=lane, K = {k0..k0+7, k0+16..k0+23}
//                          lane>=16 holds col N=lane-16, K = {k0+8..15, k0+24..31}
static __device__ __forceinline__ v16bf frag_b(const bf16* base, int ldk,
                                               int n0, int k0, int lane) {
  const bf16* p = base + (n0 + (lane & 15)) * ldk + k0 + ((lane & 16) ? 8 : 0);
  union { v16bf v; v8bf h[2]; } u;
  u.h[0] = *(const v8bf*)p;
  u.h[1] = *(const v8bf*)(p + 16);
  return u.v;
}

// LDS byte offset of a __shared__ object (generic LDS address truncates to it)
static __device__ __forceinline__ unsigned lds_off(const void* p) {
  return (unsigned)(uintptr_t)p;
}

// ---------------------------------------------------------------- router
__global__ __launch_bounds__(256) void router_kernel(
    const float* __restrict__ x, const float* __restrict__ rw,
    int* __restrict__ sel, float* __restrict__ wsum, float* __restrict__ cntf) {
  const int lane = threadIdx.x & 31;
  const int t = (blockIdx.x * 256 + threadIdx.x) >> 5;  // one wave per token
  const float* xr = x + (size_t)t * HD;
  float acc[NE];
#pragma unroll
  for (int e = 0; e < NE; ++e) acc[e] = 0.f;
  for (int j = lane; j < HD; j += 32) {
    float xv = xr[j];
#pragma unroll
    for (int e = 0; e < NE; ++e) acc[e] += xv * rw[e * HD + j];
  }
#pragma unroll
  for (int e = 0; e < NE; ++e) {
#pragma unroll
    for (int off = 16; off > 0; off >>= 1) acc[e] += __shfl_xor(acc[e], off, 32);
  }
  if (lane == 0) {
    float m = acc[0];
#pragma unroll
    for (int e = 1; e < NE; ++e) m = fmaxf(m, acc[e]);
    float p[NE];
#pragma unroll
    for (int e = 0; e < NE; ++e) p[e] = __expf(acc[e] - m);
    int i0 = 0; float b0 = p[0];
#pragma unroll
    for (int e = 1; e < NE; ++e) if (p[e] > b0) { b0 = p[e]; i0 = e; }
    int i1 = (i0 == 0) ? 1 : 0; float b1 = p[i1];
#pragma unroll
    for (int e = 0; e < NE; ++e) if (e != i0 && p[e] > b1) { b1 = p[e]; i1 = e; }
    float inv = 1.0f / (b0 + b1);
    sel[2 * t] = i0;
    sel[2 * t + 1] = i1;
    atomicAdd(&wsum[i0], b0 * inv);
    atomicAdd(&wsum[i1], b1 * inv);
    atomicAdd(&cntf[i0], 1.0f);
    atomicAdd(&cntf[i1], 1.0f);
  }
}

__global__ void finalize_kernel(const float* __restrict__ wsum,
                                const float* __restrict__ cntf,
                                float* __restrict__ meanw) {
  int e = threadIdx.x;
  if (e < NE) {
    float c = cntf[e];
    meanw[e] = (c > 0.f) ? (wsum[e] / c) : 0.f;
  }
}

__global__ __launch_bounds__(256) void compact_kernel(
    const int* __restrict__ sel, int* __restrict__ ecount, int* __restrict__ list) {
  int t = blockIdx.x * 256 + threadIdx.x;
  int e0 = sel[2 * t], e1 = sel[2 * t + 1];
  int p0 = atomicAdd(&ecount[e0], 1);
  list[e0 * T_TOK + p0] = t;
  int p1 = atomicAdd(&ecount[e1], 1);
  list[e1 * T_TOK + p1] = t;
}

__global__ __launch_bounds__(256) void cvt_x_kernel(const float* __restrict__ x,
                                                    bf16* __restrict__ xb) {
  size_t i = ((size_t)blockIdx.x * 256 + threadIdx.x) * 8;
  v4f a = *(const v4f*)(x + i);
  v4f b = *(const v4f*)(x + i + 4);
  v8bf o;
#pragma unroll
  for (int j = 0; j < 4; ++j) { o[j] = (bf16)a[j]; o[4 + j] = (bf16)b[j]; }
  *(v8bf*)(xb + i) = o;
}

// ------------------------------------------------ gate/up GEMM + SiLU (fused)
// C tile 128(M) x 64(N), K-step 64, 8 waves = 4(M)x2(N), wave tile 32x32.
// A tile (gathered token rows) staged with per-lane async global->LDS copies.
__global__ __launch_bounds__(256) void moe_gateup_kernel(
    const bf16* __restrict__ xb, const float* __restrict__ gw,
    const float* __restrict__ uw, const int* __restrict__ list,
    const int* __restrict__ ecount, bf16* __restrict__ hbuf, int e) {
  __shared__ bf16 As[128 * 64];
  __shared__ bf16 Bg[64 * 64];
  __shared__ bf16 Bu[64 * 64];

  const int cnt = ecount[e];
  const int mbase = blockIdx.y * 128;
  if (mbase >= cnt) return;
  const int nbase = blockIdx.x * 64;

  const int tid = threadIdx.x;
  const int lane = tid & 31;
  const int wave = tid >> 5;
  const int wm = (wave & 3) * 32;
  const int wn = (wave >> 2) * 32;

  // Gathered A chunks: global byte offsets vs xb + LDS byte destinations
  unsigned aoff[4];
  unsigned alds[4];
#pragma unroll
  for (int i = 0; i < 4; ++i) {
    int c = tid + 256 * i;
    int r = c >> 3;
    int col = (c & 7) * 8;
    int gr = mbase + r; if (gr >= cnt) gr = cnt - 1;
    int tok = list[e * T_TOK + gr];
    aoff[i] = (unsigned)((tok * HD + col) * 2);          // bytes from xb
    alds[i] = lds_off(As) + (unsigned)((r * 64 + col) * 2);
  }
  const float* gptr[4];
  const float* uptr[4];
  int bsidx[4];
#pragma unroll
  for (int i = 0; i < 4; ++i) {
    int c = tid + 256 * i;
    int n = c >> 4;
    int col = (c & 15) * 4;
    size_t roww = ((size_t)e * ID + nbase + n) * HD + col;
    gptr[i] = gw + roww;
    uptr[i] = uw + roww;
    bsidx[i] = n * 64 + col;
  }

  const v8f vzero = {0.f, 0.f, 0.f, 0.f, 0.f, 0.f, 0.f, 0.f};
  v8f ag[2][2], au[2][2];
#pragma unroll
  for (int a = 0; a < 2; ++a)
#pragma unroll
    for (int b = 0; b < 2; ++b) { ag[a][b] = vzero; au[a][b] = vzero; }

  for (int k = 0; k < HD; k += 64) {
    // Async copy gathered A chunks straight into LDS (ASYNCcnt path)
#pragma unroll
    for (int i = 0; i < 4; ++i) {
      unsigned voff = aoff[i] + (unsigned)(k * 2);
      asm volatile("global_load_async_to_lds_b128 %0, %1, %2"
                   :: "v"(alds[i]), "v"(voff), "s"(xb)
                   : "memory");
    }
    // Meanwhile convert weight tiles f32 -> bf16 through VGPRs
#pragma unroll
    for (int i = 0; i < 4; ++i) {
      v4f g4 = *(const v4f*)(gptr[i] + k);
      v4f u4 = *(const v4f*)(uptr[i] + k);
      v4bf gb, ub;
#pragma unroll
      for (int j = 0; j < 4; ++j) { gb[j] = (bf16)g4[j]; ub[j] = (bf16)u4[j]; }
      *(v4bf*)&Bg[bsidx[i]] = gb;
      *(v4bf*)&Bu[bsidx[i]] = ub;
      if (k + 64 < HD) {
        __builtin_prefetch(gptr[i] + k + 64, 0, 1);  // global_prefetch_b8
        __builtin_prefetch(uptr[i] + k + 64, 0, 1);
      }
    }
    asm volatile("s_wait_asynccnt 0x0" ::: "memory");
    __syncthreads();
#pragma unroll
    for (int kk = 0; kk < 64; kk += 32) {
      v16bf a0 = frag_a(As, 64, wm + 0, kk, lane);
      v16bf a1 = frag_a(As, 64, wm + 16, kk, lane);
      v16bf g0 = frag_b(Bg, 64, wn + 0, kk, lane);
      v16bf g1 = frag_b(Bg, 64, wn + 16, kk, lane);
      v16bf u0 = frag_b(Bu, 64, wn + 0, kk, lane);
      v16bf u1 = frag_b(Bu, 64, wn + 16, kk, lane);
      ag[0][0] = wmma_bf16(a0, g0, ag[0][0]);
      ag[0][1] = wmma_bf16(a0, g1, ag[0][1]);
      ag[1][0] = wmma_bf16(a1, g0, ag[1][0]);
      ag[1][1] = wmma_bf16(a1, g1, ag[1][1]);
      au[0][0] = wmma_bf16(a0, u0, au[0][0]);
      au[0][1] = wmma_bf16(a0, u1, au[0][1]);
      au[1][0] = wmma_bf16(a1, u0, au[1][0]);
      au[1][1] = wmma_bf16(a1, u1, au[1][1]);
    }
    __syncthreads();
  }

  // h = silu(gate) * up  -> bf16 hbuf (compact row index space)
  const int mrow0 = mbase + wm + ((lane & 16) ? 8 : 0);
  const int ncol = nbase + wn + (lane & 15);
#pragma unroll
  for (int mt = 0; mt < 2; ++mt)
#pragma unroll
    for (int nt = 0; nt < 2; ++nt)
#pragma unroll
      for (int v = 0; v < 8; ++v) {
        float g = ag[mt][nt][v];
        float u = au[mt][nt][v];
        float h = (g / (1.0f + __expf(-g))) * u;
        hbuf[(size_t)(mrow0 + mt * 16 + v) * ID + (ncol + nt * 16)] = (bf16)h;
      }
}

// ------------------------------------------------ down GEMM + scaled scatter
// A tile (contiguous compact rows of hbuf) staged by the Tensor Data Mover:
// one tensor_load_to_lds per K-step DMA's the 128x64 bf16 tile into LDS.
__global__ __launch_bounds__(256) void moe_down_kernel(
    const bf16* __restrict__ hbuf, const float* __restrict__ dw,
    const int* __restrict__ list, const int* __restrict__ ecount,
    const float* __restrict__ meanw, float* __restrict__ out, int e) {
  __shared__ bf16 As[128 * 64];
  __shared__ bf16 Bs[64 * 64];

  const int cnt = ecount[e];
  const int mbase = blockIdx.y * 128;
  if (mbase >= cnt) return;
  const int nbase = blockIdx.x * 64;

  const int tid = threadIdx.x;
  const int lane = tid & 31;
  const int wave = tid >> 5;
  const int wm = (wave & 3) * 32;
  const int wn = (wave >> 2) * 32;

  const unsigned asbase = lds_off(As);
  const unsigned long long abase =
      (unsigned long long)(uintptr_t)hbuf + (unsigned long long)mbase * ID * 2ull;

  const float* bptr[4];
  int bsidx[4];
#pragma unroll
  for (int i = 0; i < 4; ++i) {
    int c = tid + 256 * i;
    int n = c >> 4;
    int col = (c & 15) * 4;
    bptr[i] = dw + ((size_t)e * HD + nbase + n) * ID + col;
    bsidx[i] = n * 64 + col;
  }

  const v8f vzero = {0.f, 0.f, 0.f, 0.f, 0.f, 0.f, 0.f, 0.f};
  v8f acc[2][2];
#pragma unroll
  for (int a = 0; a < 2; ++a)
#pragma unroll
    for (int b = 0; b < 2; ++b) acc[a][b] = vzero;

  for (int k = 0; k < ID; k += 64) {
    if (wave == 0) {
      // Build Tensor DMA Descriptor (D#) for a 2D 128x64 bf16 tile.
      unsigned long long ga = abase + (unsigned long long)(k * 2);
      u32x4 g0;
      g0[0] = 1u;                                   // count=1, load, no gather
      g0[1] = asbase;                               // lds_addr
      g0[2] = (unsigned)ga;                         // global_addr[31:0]
      g0[3] = (unsigned)(ga >> 32) | (2u << 30);    // global_addr[56:32] | type=2
      u32x8 g1;
      g1[0] = 1u << 16;                             // data_size=1 (2 bytes)
      g1[1] = ((unsigned)ID & 0xffffu) << 16;       // tensor_dim0 lo16
      g1[2] = ((unsigned)ID >> 16) |
              (((unsigned)T_TOK & 0xffffu) << 16);  // dim0 hi16 | tensor_dim1 lo16
      g1[3] = ((unsigned)T_TOK >> 16) | (64u << 16);// dim1 hi16 | tile_dim0=64
      g1[4] = 128u;                                 // tile_dim1=128, tile_dim2=0
      g1[5] = (unsigned)ID;                         // tensor_dim0_stride lo32
      g1[6] = 0u;
      g1[7] = 0u;
      asm volatile("tensor_load_to_lds %0, %1\n\t"
                   "s_wait_tensorcnt 0x0"
                   :: "s"(g0), "s"(g1)
                   : "memory");
    }
    // All threads convert the down-weight tile f32 -> bf16 while the TDM runs
#pragma unroll
    for (int i = 0; i < 4; ++i) {
      v4f b4 = *(const v4f*)(bptr[i] + k);
      v4bf bb;
#pragma unroll
      for (int j = 0; j < 4; ++j) bb[j] = (bf16)b4[j];
      *(v4bf*)&Bs[bsidx[i]] = bb;
      if (k + 64 < ID) __builtin_prefetch(bptr[i] + k + 64, 0, 1);
    }
    __syncthreads();
#pragma unroll
    for (int kk = 0; kk < 64; kk += 32) {
      v16bf a0 = frag_a(As, 64, wm + 0, kk, lane);
      v16bf a1 = frag_a(As, 64, wm + 16, kk, lane);
      v16bf b0 = frag_b(Bs, 64, wn + 0, kk, lane);
      v16bf b1 = frag_b(Bs, 64, wn + 16, kk, lane);
      acc[0][0] = wmma_bf16(a0, b0, acc[0][0]);
      acc[0][1] = wmma_bf16(a0, b1, acc[0][1]);
      acc[1][0] = wmma_bf16(a1, b0, acc[1][0]);
      acc[1][1] = wmma_bf16(a1, b1, acc[1][1]);
    }
    __syncthreads();
  }

  const float scale = meanw[e];
  const int mrow0 = mbase + wm + ((lane & 16) ? 8 : 0);
  const int ncol = nbase + wn + (lane & 15);
#pragma unroll
  for (int mt = 0; mt < 2; ++mt)
#pragma unroll
    for (int nt = 0; nt < 2; ++nt)
#pragma unroll
      for (int v = 0; v < 8; ++v) {
        int m = mrow0 + mt * 16 + v;
        if (m < cnt) {
          int tok = list[e * T_TOK + m];
          atomicAdd(&out[(size_t)tok * HD + (ncol + nt * 16)],
                    acc[mt][nt][v] * scale);  // global_atomic_add_f32
        }
      }
}

// ---------------------------------------------------------------- launcher
extern "C" void kernel_launch(void* const* d_in, const int* in_sizes, int n_in,
                              void* d_out, int out_size, void* d_ws, size_t ws_size,
                              hipStream_t stream) {
  const float* x  = (const float*)d_in[0];
  const float* rw = (const float*)d_in[1];
  const float* gw = (const float*)d_in[2];
  const float* uw = (const float*)d_in[3];
  const float* dw = (const float*)d_in[4];
  float* out = (float*)d_out;

  char* ws = (char*)d_ws;
  float* wsum  = (float*)ws;          // 8 f32
  float* cntf  = wsum + NE;           // 8 f32
  float* meanw = cntf + NE;           // 8 f32
  int* ecount  = (int*)(meanw + NE);  // 8 i32   (first 128 bytes total)
  int* sel     = ecount + NE;         // T*2 i32
  int* list    = sel + 2 * T_TOK;     // NE*T i32
  uintptr_t p = (uintptr_t)(list + NE * T_TOK);
  p = (p + 255) & ~(uintptr_t)255;
  bf16* xb = (bf16*)p;                // T*H bf16   (16.8 MB)
  p += (size_t)T_TOK * HD * sizeof(bf16);
  bf16* hbuf = (bf16*)p;              // T*I bf16   (67 MB)

  hipMemsetAsync(d_ws, 0, 128, stream);
  hipMemsetAsync(d_out, 0, (size_t)out_size * sizeof(float), stream);

  cvt_x_kernel<<<(T_TOK * HD) / (256 * 8), 256, 0, stream>>>(x, xb);
  router_kernel<<<T_TOK / 8, 256, 0, stream>>>(x, rw, sel, wsum, cntf);
  finalize_kernel<<<1, 32, 0, stream>>>(wsum, cntf, meanw);
  compact_kernel<<<T_TOK / 256, 256, 0, stream>>>(sel, ecount, list);

  dim3 gA(ID / 64, T_TOK / 128);
  dim3 gB(HD / 64, T_TOK / 128);
  for (int e = 0; e < NE; ++e) {
    moe_gateup_kernel<<<gA, 256, 0, stream>>>(xb, gw, uw, list, ecount, hbuf, e);
    moe_down_kernel<<<gB, 256, 0, stream>>>(hbuf, dw, list, ecount, meanw, out, e);
  }
}
